// model_8589935220
// MI455X (gfx1250) — matrix operand order repaired and checked by reading, not compiled, not run on hardware
//
#include <hip/hip_runtime.h>
#include <math.h>

#define NN 50000
#define NE 800000

typedef __attribute__((ext_vector_type(16))) _Float16 v16h;
typedef __attribute__((ext_vector_type(2)))  __fp16   v2fp;   // cvt_pkrtz return type
typedef __attribute__((ext_vector_type(8)))  float    v8f;

union Frag16 { v16h v; v2fp p[8]; };

// ---------------------------------------------------------------- zero
__global__ void zero_kernel(float* __restrict__ p, int n) {
    int i = blockIdx.x * blockDim.x + threadIdx.x;
    if (i < n) p[i] = 0.0f;
}

// ------------------------------------------------- edge scatter-aggregate
// aggr[dst[e], c] += ew[e] * h[src[e], c]   (one thread per (edge, channel))
__global__ void scatter_kernel(const float* __restrict__ h,
                               const int*   __restrict__ src,
                               const int*   __restrict__ dst,
                               const float* __restrict__ ew,
                               float* __restrict__ aggr, int d) {
    int i = blockIdx.x * blockDim.x + threadIdx.x;
    int total = NE * d;
    if (i >= total) return;
    int e = i / d;
    int c = i - e * d;
    float v = ew[e] * h[src[e] * d + c];
    // relaxed, device scope -> native global_atomic_add_f32 (L2 atomic unit)
    __hip_atomic_fetch_add(&aggr[dst[e] * d + c], v,
                           __ATOMIC_RELAXED, __HIP_MEMORY_SCOPE_AGENT);
}

// pack 16 K-contiguous fp32 values (two 8-float runs) into a WMMA A/B v16h
__device__ __forceinline__ v16h pack16(const float4 f0, const float4 f1,
                                       const float4 f2, const float4 f3) {
    Frag16 fr;
    fr.p[0] = __builtin_amdgcn_cvt_pkrtz(f0.x, f0.y);
    fr.p[1] = __builtin_amdgcn_cvt_pkrtz(f0.z, f0.w);
    fr.p[2] = __builtin_amdgcn_cvt_pkrtz(f1.x, f1.y);
    fr.p[3] = __builtin_amdgcn_cvt_pkrtz(f1.z, f1.w);
    fr.p[4] = __builtin_amdgcn_cvt_pkrtz(f2.x, f2.y);
    fr.p[5] = __builtin_amdgcn_cvt_pkrtz(f2.z, f2.w);
    fr.p[6] = __builtin_amdgcn_cvt_pkrtz(f3.x, f3.y);
    fr.p[7] = __builtin_amdgcn_cvt_pkrtz(f3.z, f3.w);
    return fr.v;
}

// ------------------------------------------------------- fused WMMA GEMM
// out = act( aggr @ Wrel + brel + h @ Wroot )
// One wave32 computes a 16 x (NT*16) tile. ACT: 0=none, 1=relu, 2=sigmoid.
// LDS holds the weights pre-swizzled into per-lane fragment order so each
// lane fetches its whole 16-half B fragment as 32 contiguous bytes.
template <int ACT, int NT, int KPAD>
__global__ void gemm_wmma_kernel(const float* __restrict__ aggr,
                                 const float* __restrict__ h,
                                 const float* __restrict__ Wrel,
                                 const float* __restrict__ brel,
                                 const float* __restrict__ Wroot,
                                 float* __restrict__ out,
                                 int d_in, int d_out) {
    constexpr int KSTEPS = KPAD / 32;
    constexpr int NHALF  = KSTEPS * NT * 512;   // halves per weight matrix
    __shared__ __align__(32) _Float16 sWrel[NHALF];
    __shared__ __align__(32) _Float16 sWroot[NHALF];

    const int tid = threadIdx.x;

    // stage weights into LDS, swizzled to fragment order:
    // idx = ((ks*NT + nt)*32 + lane)*16 + e
    //   -> K = ks*32 + 8*(lane>>4) + (e&7) + 16*(e>>3),  N = nt*16 + (lane&15)
    for (int idx = tid; idx < NHALF; idx += blockDim.x) {
        int e   = idx & 15;
        int L   = (idx >> 4) & 31;
        int blk = idx >> 9;
        int nt  = blk % NT;
        int ks  = blk / NT;
        int k   = ks * 32 + ((L >> 4) << 3) + (e & 7) + ((e >> 3) << 4);
        int n   = nt * 16 + (L & 15);
        bool ok = (k < d_in) && (n < d_out);
        sWrel[idx]  = (_Float16)(ok ? Wrel[k * d_out + n]  : 0.0f);
        sWroot[idx] = (_Float16)(ok ? Wroot[k * d_out + n] : 0.0f);
    }
    __syncthreads();

    const int wave = tid >> 5;
    const int lane = tid & 31;
    const int hi   = lane >> 4;
    const int col  = lane & 15;

    const int tile = blockIdx.x * (blockDim.x >> 5) + wave;
    const int arow = tile * 16 + col;          // A fragment row: M = lane&15
    const bool arow_ok = (arow < NN);
    const float* agg_row = aggr + (size_t)(arow_ok ? arow : 0) * d_in;
    const float* h_row   = h    + (size_t)(arow_ok ? arow : 0) * d_in;

    // init accumulators with bias (C/D layout: col = lane&15 for all 8 rows)
    v8f acc[NT];
#pragma unroll
    for (int nt = 0; nt < NT; ++nt) {
        int n = nt * 16 + col;
        float b = (n < d_out) ? brel[n] : 0.0f;
#pragma unroll
        for (int v = 0; v < 8; ++v) acc[nt][v] = b;
    }

#pragma unroll
    for (int ks = 0; ks < KSTEPS; ++ks) {
        const int kb = ks * 32;
        const int k0 = kb + 8 * hi;        // A elements 0..7 : K = k0..k0+7
        const int k1 = kb + 16 + 8 * hi;   // A elements 8..15: K = k1..k1+7

        v16h a_rel, a_root;
        if (KPAD == 64) {
            // d_in == 64: fully in-range, 32B-aligned -> global_load_b128 x2
            const float4* ar  = (const float4*)(agg_row + k0);
            const float4* hr  = (const float4*)(h_row + k0);
            const float4* ar2 = (const float4*)(agg_row + k1);
            const float4* hr2 = (const float4*)(h_row + k1);
            a_rel  = pack16(ar[0], ar[1], ar2[0], ar2[1]);
            a_root = pack16(hr[0], hr[1], hr2[0], hr2[1]);
        } else {
            // d_in == 13: guarded scalar gather, then pack
            float ta[16], th[16];
#pragma unroll
            for (int j = 0; j < 8; ++j) {
                bool ok0 = arow_ok && (k0 + j) < d_in;
                bool ok1 = arow_ok && (k1 + j) < d_in;
                ta[j]     = ok0 ? agg_row[k0 + j] : 0.0f;
                ta[8 + j] = ok1 ? agg_row[k1 + j] : 0.0f;
                th[j]     = ok0 ? h_row[k0 + j] : 0.0f;
                th[8 + j] = ok1 ? h_row[k1 + j] : 0.0f;
            }
            Frag16 fa, fh;
#pragma unroll
            for (int j = 0; j < 8; ++j) {
                fa.p[j] = __builtin_amdgcn_cvt_pkrtz(ta[2 * j], ta[2 * j + 1]);
                fh.p[j] = __builtin_amdgcn_cvt_pkrtz(th[2 * j], th[2 * j + 1]);
            }
            a_rel  = fa.v;
            a_root = fh.v;
        }

#pragma unroll
        for (int nt = 0; nt < NT; ++nt) {
            // B fragment: 32 contiguous bytes per lane -> ds_load_b128 x2
            const int foff = ((ks * NT + nt) * 32 + lane) * 16;
            v16h b_rel  = *(const v16h*)(sWrel + foff);
            v16h b_root = *(const v16h*)(sWroot + foff);
            acc[nt] = __builtin_amdgcn_wmma_f32_16x16x32_f16(
                false, a_rel, false, b_rel, (short)0, acc[nt], false, false);
            acc[nt] = __builtin_amdgcn_wmma_f32_16x16x32_f16(
                false, a_root, false, b_root, (short)0, acc[nt], false, false);
        }
    }

    // store D: element v -> row = tile*16 + v + 8*hi, col = nt*16 + (lane&15)
#pragma unroll
    for (int nt = 0; nt < NT; ++nt) {
        int n = nt * 16 + col;
        if (n >= d_out) continue;
#pragma unroll
        for (int v = 0; v < 8; ++v) {
            int row = tile * 16 + v + 8 * hi;
            if (row < NN) {
                float val = acc[nt][v];
                if (ACT == 1) val = val > 0.0f ? val : 0.0f;
                if (ACT == 2) val = 1.0f / (1.0f + __expf(-val));
                out[(size_t)row * d_out + n] = val;
            }
        }
    }
}

// ---------------------------------------------------------------- driver
static inline void run_layer(const float* hin, int d_in,
                             const int* src, const int* dst, const float* ew,
                             const float* Wr, const float* br, const float* Ws,
                             float* aggr, float* hout, int layer,
                             hipStream_t stream) {
    int nz = NN * d_in;
    zero_kernel<<<(nz + 255) / 256, 256, 0, stream>>>(aggr, nz);
    int ns = NE * d_in;
    scatter_kernel<<<(ns + 255) / 256, 256, 0, stream>>>(hin, src, dst, ew, aggr, d_in);
    const int tiles  = (NN + 15) / 16;        // 3125
    const int blocks = (tiles + 7) / 8;       // 391 (8 waves / block)
    if (layer == 0) {
        gemm_wmma_kernel<1, 4, 32><<<blocks, 256, 0, stream>>>(
            aggr, hin, Wr, br, Ws, hout, d_in, 64);
    } else if (layer < 4) {
        gemm_wmma_kernel<1, 4, 64><<<blocks, 256, 0, stream>>>(
            aggr, hin, Wr, br, Ws, hout, d_in, 64);
    } else {
        gemm_wmma_kernel<2, 1, 64><<<blocks, 256, 0, stream>>>(
            aggr, hin, Wr, br, Ws, hout, d_in, 1);
    }
}

extern "C" void kernel_launch(void* const* d_in, const int* in_sizes, int n_in,
                              void* d_out, int out_size, void* d_ws, size_t ws_size,
                              hipStream_t stream) {
    const float* x   = (const float*)d_in[0];
    const int*   ei  = (const int*)d_in[1];
    const float* ew  = (const float*)d_in[2];
    const int* src = ei;            // edge_index[0]
    const int* dst = ei + NE;       // edge_index[1]

    const float *Wr[5], *br[5], *Ws[5];
    for (int i = 0; i < 5; ++i) {
        Wr[i] = (const float*)d_in[3 + 3 * i];
        br[i] = (const float*)d_in[4 + 3 * i];
        Ws[i] = (const float*)d_in[5 + 3 * i];
    }

    float* ws   = (float*)d_ws;
    float* aggr = ws;                       // NN*64 floats
    float* hA   = ws + (size_t)NN * 64;     // NN*64
    float* hB   = hA + (size_t)NN * 64;     // NN*64

    // layer 0: x[50000,13] -> hA[50000,64]
    run_layer(x, 13, src, dst, ew, Wr[0], br[0], Ws[0], aggr, hA, 0, stream);
    // layers 1..3: ping-pong hA <-> hB
    run_layer(hA, 64, src, dst, ew, Wr[1], br[1], Ws[1], aggr, hB, 1, stream);
    run_layer(hB, 64, src, dst, ew, Wr[2], br[2], Ws[2], aggr, hA, 2, stream);
    run_layer(hA, 64, src, dst, ew, Wr[3], br[3], Ws[3], aggr, hB, 3, stream);
    // layer 4: -> sigmoid -> d_out [50000,1]
    run_layer(hB, 64, src, dst, ew, Wr[4], br[4], Ws[4], aggr, (float*)d_out, 4, stream);
}
